// MaskedRoIExtractor_64793876627955
// MI455X (gfx1250) — matrix operand order repaired and checked by reading, not compiled, not run on hardware
//
#include <hip/hip_runtime.h>
#include <hip/hip_bf16.h>

// ---------------------------------------------------------------------------
// MaskedRoIExtractor for MI455X (gfx1250, wave32, WMMA)
//   Stage 1: 3x3 conv (CIN=256 -> COUT=128) as implicit GEMM via
//            v_wmma_f32_16x16x32_bf16 with a 2x2 register tile per wave
//            (4 WMMAs per A/B fragment pair -> 2.0 b128-loads per WMMA),
//            folded bias/BN/ReLU, NHWC output.
//   Stage 2: masked bilinear RoI sampling (gather, coalesced NHWC reads).
// ---------------------------------------------------------------------------

#define HF     256
#define WF     256
#define CIN_   256
#define COUT_  128
#define BATCH_ 4
#define OUTH   8
#define OUTW   32
#define NROIS  128
#define NFRAG  (9 * 8 * 8)       // taps * cin-chunks * cout-tiles = 576
#define FRAG_ELEMS 512           // 32 lanes * 16 bf16

typedef __attribute__((ext_vector_type(16))) __bf16 v16bf;
typedef __attribute__((ext_vector_type(8)))  float  v8f;
typedef __attribute__((ext_vector_type(4)))  unsigned int u32x4;

struct Pair16B { u32x4 a, b; };   // 32 bytes == one v16bf fragment per lane

static __device__ __forceinline__ unsigned short f32_to_bf16_rne(float f) {
    unsigned int u = __float_as_uint(f);
    u += 0x7fffu + ((u >> 16) & 1u);   // round-to-nearest-even
    return (unsigned short)(u >> 16);
}

static __device__ __forceinline__ v16bf load_frag2(const unsigned short* p0,
                                                   const unsigned short* p1) {
    Pair16B s;
    s.a = *(const u32x4*)p0;   // 8 bf16 (16B)
    s.b = *(const u32x4*)p1;   // 8 bf16 (16B)
    return __builtin_bit_cast(v16bf, s);
}

// ---------------------------------------------------------------------------
// Prepack: fold conv bias + BN into per-channel scale/bias
// ---------------------------------------------------------------------------
__global__ void prep_scale_bias(const float* __restrict__ convb,
                                const float* __restrict__ gamma,
                                const float* __restrict__ beta,
                                const float* __restrict__ mean,
                                const float* __restrict__ var,
                                float* __restrict__ scale,
                                float* __restrict__ bias) {
    int c = threadIdx.x;
    if (c < COUT_) {
        float s = gamma[c] * rsqrtf(var[c] + 1e-5f);
        scale[c] = s;
        bias[c]  = (convb[c] - mean[c]) * s + beta[c];
    }
}

// ---------------------------------------------------------------------------
// Prepack: weights OIHW f32 -> bf16 B-fragments in native WMMA lane layout.
// B (KxN = 32 cin x 16 cout): lane<16 holds column N=lane, K={0..7,16..23};
// lane>=16 holds column N=lane-16, K={8..15,24..31}. 32B contiguous per lane.
// ---------------------------------------------------------------------------
__global__ void pack_weights(const float* __restrict__ w,
                             unsigned short* __restrict__ wfrag) {
    int g = blockIdx.x * blockDim.x + threadIdx.x;
    if (g >= NFRAG * FRAG_ELEMS) return;
    int e    = g & 15;
    int lane = (g >> 4) & 31;
    int frag = g >> 9;
    int nt   = frag & 7;
    int kc   = (frag >> 3) & 7;
    int tap  = frag >> 6;                 // ky*3 + kx
    int n    = lane & 15;
    int k    = (e & 7) + ((lane >> 4) << 3) + ((e >> 3) << 4);
    int cout = nt * 16 + n;
    int cin  = kc * 32 + k;
    float v  = w[((size_t)cout * CIN_ + cin) * 9 + tap];
    wfrag[g] = f32_to_bf16_rne(v);
}

// ---------------------------------------------------------------------------
// Conv 3x3 + BN + ReLU, implicit GEMM via WMMA bf16.
// Block: one (b,y) x 32 pixels x 128 cout. 128 threads = 4 wave32.
// Each wave: 2 M-tiles x 2 N-tiles (4 accumulators) -> each A/B fragment
// pair feeds 4 WMMAs.
// LDS: 3 rows x 34 cols x 256 cin, bf16 (52 KB) with on-the-fly f32->bf16.
// ---------------------------------------------------------------------------
__launch_bounds__(128)
__global__ void conv_bn_relu_wmma(const float* __restrict__ feats,          // NCHW f32
                                  const unsigned short* __restrict__ wfrag, // packed bf16
                                  const float* __restrict__ scale,
                                  const float* __restrict__ bias,
                                  float* __restrict__ outNHWC) {
    __shared__ unsigned short sIn[3 * 34 * CIN_];   // [row][xi][cin]

    const int x0  = blockIdx.x * 32;   // pixel tile
    const int y   = blockIdx.y;
    const int b   = blockIdx.z;
    const int tid = threadIdx.x;

    // ---- stage input halo tile into LDS as bf16 (zero-pad borders) ----
    for (int idx = tid; idx < 3 * 34 * CIN_; idx += 128) {
        int cin = idx / 102;
        int rem = idx - cin * 102;
        int row = rem / 34;
        int xi  = rem - row * 34;
        int yy  = y + row - 1;
        int xx  = x0 + xi - 1;
        float v = 0.0f;
        if (yy >= 0 && yy < HF && xx >= 0 && xx < WF)
            v = feats[(((size_t)b * CIN_ + cin) * HF + yy) * WF + xx];
        sIn[(row * 34 + xi) * CIN_ + cin] = f32_to_bf16_rne(v);
    }
    __syncthreads();

    const int wv     = tid >> 5;           // 0..3
    const int lane   = tid & 31;
    const int lane15 = lane & 15;
    const int hi     = lane >> 4;          // half-wave select
    const int nt0    = wv * 2;             // cout tiles handled by this wave
    const int nt1    = nt0 + 1;

    v8f acc00 = {};   // (M-tile 0, N-tile nt0)
    v8f acc01 = {};   // (M-tile 0, N-tile nt1)
    v8f acc10 = {};   // (M-tile 1, N-tile nt0)
    v8f acc11 = {};   // (M-tile 1, N-tile nt1)

#pragma unroll
    for (int tap = 0; tap < 9; ++tap) {
        const int dy = tap / 3;            // constant after unroll
        const int dx = tap - dy * 3;
        // A rows for pixel tiles 0 (x0..x0+15) and 1 (x0+16..x0+31)
        const unsigned short* arow0 = &sIn[(dy * 34 + lane15 + dx) * CIN_];
#pragma unroll
        for (int kc = 0; kc < 8; ++kc) {
            // A fragments (16 pixels x 32 cin), documented 16-bit A layout:
            // lane<16: K={0..7,16..23}; lane>=16: K={8..15,24..31}
            const unsigned short* ap0 = arow0 + kc * 32 + hi * 8;
            const unsigned short* ap1 = ap0 + 16 * CIN_;   // pixel +16
            v16bf a0 = load_frag2(ap0, ap0 + 16);
            v16bf a1 = load_frag2(ap1, ap1 + 16);

            // B fragments (32 cin x 16 cout), pre-swizzled: 32B/lane contiguous
            const int fb = (tap * 8 + kc) * 8 + nt0;
            const unsigned short* bp0 = wfrag + ((size_t)fb * 32 + lane) * 16;
            const unsigned short* bp1 = bp0 + 32 * 16;     // next N-tile fragment
            v16bf b0 = load_frag2(bp0, bp0 + 8);
            v16bf b1 = load_frag2(bp1, bp1 + 8);

            acc00 = __builtin_amdgcn_wmma_f32_16x16x32_bf16(false, a0, false, b0,
                                                            (short)0, acc00, false, false);
            acc01 = __builtin_amdgcn_wmma_f32_16x16x32_bf16(false, a0, false, b1,
                                                            (short)0, acc01, false, false);
            acc10 = __builtin_amdgcn_wmma_f32_16x16x32_bf16(false, a1, false, b0,
                                                            (short)0, acc10, false, false);
            acc11 = __builtin_amdgcn_wmma_f32_16x16x32_bf16(false, a1, false, b1,
                                                            (short)0, acc11, false, false);
        }
    }

    // ---- epilogue: scale/bias/ReLU, store NHWC ----
    // D layout: lane<16 -> (M=r, N=lane15); lane>=16 -> (M=r+8, N=lane15)
    const int c0 = nt0 * 16 + lane15;
    const int c1 = nt1 * 16 + lane15;
    const float s0 = scale[c0], o0 = bias[c0];
    const float s1 = scale[c1], o1 = bias[c1];
    const int px0 = hi * 8;
    float* orow = outNHWC + (((size_t)b * HF + y) * WF + x0) * COUT_;
#pragma unroll
    for (int r = 0; r < 8; ++r) {
        const int p0 = px0 + r;         // pixel in tile 0
        const int p1 = p0 + 16;         // pixel in tile 1
        orow[(size_t)p0 * COUT_ + c0] = fmaxf(acc00[r] * s0 + o0, 0.0f);
        orow[(size_t)p0 * COUT_ + c1] = fmaxf(acc01[r] * s1 + o1, 0.0f);
        orow[(size_t)p1 * COUT_ + c0] = fmaxf(acc10[r] * s0 + o0, 0.0f);
        orow[(size_t)p1 * COUT_ + c1] = fmaxf(acc11[r] * s1 + o1, 0.0f);
    }
}

// ---------------------------------------------------------------------------
// Masked bilinear RoI extraction. feat is NHWC -> 128-ch contiguous gathers.
// Grid: (roi, oy); thread: ox = t>>3, cout chunk of 16 = (t&7)*16.
// ---------------------------------------------------------------------------
__launch_bounds__(256)
__global__ void roi_extract(const float* __restrict__ feat,   // [B,H,W,COUT]
                            const float* __restrict__ rois,   // [NROIS,5]
                            const float* __restrict__ masks,  // [NROIS,H,W]
                            float* __restrict__ out) {        // [NROIS,COUT,OUTH,OUTW]
    const int roi = blockIdx.x;
    const int oy  = blockIdx.y;
    const int tid = threadIdx.x;
    const int ox  = tid >> 3;
    const int c16 = (tid & 7) * 16;

    const float* rr = rois + (size_t)roi * 5;
    const int bi = (int)rr[0];
    const int bx0 = (int)(rr[1] * 0.25f - 1.0f);
    const int by0 = (int)(rr[2] * 0.25f - 1.0f);
    const int bx1 = (int)(rr[3] * 0.25f + 1.0f);
    const int by1 = (int)(rr[4] * 0.25f + 1.0f);
    const int l = min(max(bx0, 0), WF);
    const int t = min(max(by0, 0), HF);
    const int r = min(max(bx1, 0), WF);
    const int b = min(max(by1, 0), HF);
    const bool valid = (r > l) && (b > t);

    const float ch = (float)(b - t);
    const float cw = (float)(r - l);
    const bool  tr = ch > 1.5f * cw;
    const float fi = (float)oy + 0.5f;
    const float fj = (float)ox + 0.5f;
    float V = tr ? (fj * ch / OUTW - 0.5f) : (fi * ch / OUTH - 0.5f);
    float U = tr ? (fi * cw / OUTH - 0.5f) : (fj * cw / OUTW - 0.5f);
    V = fmaxf(V, 0.0f);
    U = fmaxf(U, 0.0f);
    const int chi = max(b - t, 1);
    const int cwi = max(r - l, 1);
    const float vf = floorf(V);
    const float uf = floorf(U);
    const int y0 = min((int)vf, chi - 1);
    const int y1 = min(y0 + 1, chi - 1);
    const int x0 = min((int)uf, cwi - 1);
    const int x1 = min(x0 + 1, cwi - 1);
    const float wy = V - vf;
    const float wx = U - uf;
    const int ay0 = min(max(t + y0, 0), HF - 1);
    const int ay1 = min(max(t + y1, 0), HF - 1);
    const int ax0 = min(max(l + x0, 0), WF - 1);
    const int ax1 = min(max(l + x1, 0), WF - 1);

    const float* mrow = masks + (size_t)roi * HF * WF;
    const float w00 = (1.0f - wy) * (1.0f - wx) * mrow[ay0 * WF + ax0];
    const float w01 = (1.0f - wy) * wx          * mrow[ay0 * WF + ax1];
    const float w10 = wy          * (1.0f - wx) * mrow[ay1 * WF + ax0];
    const float w11 = wy          * wx          * mrow[ay1 * WF + ax1];

    const float* base = feat + ((size_t)bi * HF * WF) * COUT_ + c16;
    const float4* p00 = (const float4*)(base + ((size_t)ay0 * WF + ax0) * COUT_);
    const float4* p01 = (const float4*)(base + ((size_t)ay0 * WF + ax1) * COUT_);
    const float4* p10 = (const float4*)(base + ((size_t)ay1 * WF + ax0) * COUT_);
    const float4* p11 = (const float4*)(base + ((size_t)ay1 * WF + ax1) * COUT_);

    float* op = out + (((size_t)roi * COUT_ + c16) * OUTH + oy) * OUTW + ox;
#pragma unroll
    for (int q = 0; q < 4; ++q) {
        float4 a0 = p00[q], a1 = p01[q], a2 = p10[q], a3 = p11[q];
        float v0 = w00 * a0.x + w01 * a1.x + w10 * a2.x + w11 * a3.x;
        float v1 = w00 * a0.y + w01 * a1.y + w10 * a2.y + w11 * a3.y;
        float v2 = w00 * a0.z + w01 * a1.z + w10 * a2.z + w11 * a3.z;
        float v3 = w00 * a0.w + w01 * a1.w + w10 * a2.w + w11 * a3.w;
        if (!valid) { v0 = v1 = v2 = v3 = 0.0f; }
        op[(size_t)(q * 4 + 0) * (OUTH * OUTW)] = v0;
        op[(size_t)(q * 4 + 1) * (OUTH * OUTW)] = v1;
        op[(size_t)(q * 4 + 2) * (OUTH * OUTW)] = v2;
        op[(size_t)(q * 4 + 3) * (OUTH * OUTW)] = v3;
    }
}

// ---------------------------------------------------------------------------
extern "C" void kernel_launch(void* const* d_in, const int* in_sizes, int n_in,
                              void* d_out, int out_size, void* d_ws, size_t ws_size,
                              hipStream_t stream) {
    const float* feats = (const float*)d_in[0];
    const float* rois  = (const float*)d_in[1];
    const float* masks = (const float*)d_in[2];
    const float* convw = (const float*)d_in[3];
    const float* convb = (const float*)d_in[4];
    const float* gamma = (const float*)d_in[5];
    const float* beta  = (const float*)d_in[6];
    const float* mean  = (const float*)d_in[7];
    const float* var   = (const float*)d_in[8];
    float* out = (float*)d_out;

    char* ws = (char*)d_ws;
    float* convOut = (float*)ws;                                       // NHWC f32
    const size_t convBytes = (size_t)BATCH_ * HF * WF * COUT_ * sizeof(float);
    unsigned short* wfrag = (unsigned short*)(ws + convBytes);         // packed bf16
    const size_t wfragBytes = (size_t)NFRAG * FRAG_ELEMS * sizeof(unsigned short);
    float* scale = (float*)(ws + convBytes + wfragBytes);
    float* bias  = scale + COUT_;

    prep_scale_bias<<<1, 128, 0, stream>>>(convb, gamma, beta, mean, var, scale, bias);
    pack_weights<<<(NFRAG * FRAG_ELEMS + 255) / 256, 256, 0, stream>>>(convw, wfrag);
    conv_bn_relu_wmma<<<dim3(WF / 32, HF, BATCH_), 128, 0, stream>>>(
        feats, wfrag, scale, bias, convOut);
    roi_extract<<<dim3(NROIS, OUTH), 256, 0, stream>>>(convOut, rois, masks, out);
}
